// FrameQuerySwin_VIS111_87316685128170
// MI455X (gfx1250) — compile-verified, matrix operand order checked by reading, not run on hardware
//
#include <hip/hip_runtime.h>
#include <cstdint>
#include <cstddef>

#define T_FRAMES 48
#define FQN 100
#define DMODEL 512
#define NHEAD 8
#define DFF 2048
#define NLAYER 6
#define NW 6            // T / WSZ
#define S_WIN 800       // WSZ * FQ
#define ROWS 9600       // B * T * FQ
#define BHN 96          // B * NW * H
#define SPAD 896        // 7 * 128 padded q rows per (window,head)

typedef __attribute__((ext_vector_type(16))) __bf16 v16bf;
typedef __attribute__((ext_vector_type(2)))  __bf16 v2bf;
typedef __attribute__((ext_vector_type(8)))  float  v8f;

union FragB { v16bf v; uint4 q[2]; unsigned short s[16]; };

// native bf16 convert (v_cvt_bf16_f32 / v_cvt_pk_bf16_f32 on CDNA5)
__device__ inline unsigned short f2bf(float f) {
  return __builtin_bit_cast(unsigned short, (__bf16)f);
}
__device__ inline unsigned pack2(float a, float b) {
#if __has_builtin(__builtin_amdgcn_cvt_pk_bf16_f32)
  v2bf r = __builtin_amdgcn_cvt_pk_bf16_f32(a, b);
  return __builtin_bit_cast(unsigned, r);
#else
  return (unsigned)f2bf(a) | ((unsigned)f2bf(b) << 16);
#endif
}

__device__ inline v8f wmma_bf(const FragB& a, const FragB& b, v8f c) {
  return __builtin_amdgcn_wmma_f32_16x16x32_bf16(false, a.v, false, b.v,
                                                 (short)0, c, false, false);
}

// CDNA5 async global->LDS copy (16B per lane), tracked by ASYNCcnt.
__device__ inline void async_b128(const void* g, void* lds) {
  asm volatile("global_load_async_to_lds_b128 %0, %1, off"
               :: "v"((unsigned)(uintptr_t)lds),
                  "v"((unsigned long long)(uintptr_t)g)
               : "memory");
}
__device__ inline void wait_async0() {
  asm volatile("s_wait_asynccnt 0x0" ::: "memory");
}

// ---------------------------------------------------------------------------
// Tiled GEMM, bf16 operands: C[M,N] = A[M,K] * W[N,K]^T + bias.
// Block tile 128x128x32, 8 waves as 4(M)x2(N), 2x4 WMMA accumulators/wave.
// A/W tiles streamed via global_load_async_to_lds_b128 with ping-pong LDS.
// MODE 0: store f32. MODE 1: relu -> bf16. MODE 2: QKV-pack bf16 per-head
// (Q x0.125). MODE 3: store f32 + bf16.
// ---------------------------------------------------------------------------
#define BM 128
#define BN 128
#define BKK 32
#define LDK 40          // padded LDS row stride (elements) -> bank-conflict free

template<int MODE>
__global__ __launch_bounds__(256)
void gemm_bf16(const unsigned short* __restrict__ A,
               const unsigned short* __restrict__ W,
               const float* __restrict__ bias,
               float* __restrict__ Cf, unsigned short* __restrict__ Cb,
               unsigned short* __restrict__ Qb, unsigned short* __restrict__ Kb,
               unsigned short* __restrict__ Vb, int M, int N, int K)
{
  __shared__ unsigned short As[2][BM * LDK];
  __shared__ unsigned short Ws[2][BN * LDK];
  const int tid  = threadIdx.x;
  const int lane = tid & 31, wave = tid >> 5;
  const int wm = wave & 3, wn = wave >> 2;
  const int m0 = blockIdx.y * BM, n0 = blockIdx.x * BN;
  const int r16 = lane & 15, cb = (lane >> 4) << 3;   // cb in {0,8}

  v8f acc[2][4];
  #pragma unroll
  for (int a = 0; a < 2; a++)
    #pragma unroll
    for (int b = 0; b < 4; b++)
      #pragma unroll
      for (int j = 0; j < 8; j++) acc[a][b][j] = 0.f;

  // each thread issues 2 x 16B async copies per matrix per k-step
  auto issue = [&](int buf, int k0) {
    #pragma unroll
    for (int i = 0; i < 2; i++) {
      int idx = tid + (i << 8);
      int r = idx >> 2, ch = (idx & 3) << 3;
      async_b128(A + (size_t)(m0 + r) * K + k0 + ch, &As[buf][r * LDK + ch]);
      async_b128(W + (size_t)(n0 + r) * K + k0 + ch, &Ws[buf][r * LDK + ch]);
    }
  };

  const int nk = K / BKK;
  issue(0, 0);
  for (int s = 0; s < nk; s++) {
    const int buf = s & 1;
    wait_async0();
    __syncthreads();                       // tile[buf] complete for all waves
    if (s + 1 < nk) issue(buf ^ 1, (s + 1) * BKK);

    FragB af[2], bf_[4];
    #pragma unroll
    for (int mt = 0; mt < 2; mt++) {
      int r = (wm << 5) + (mt << 4) + r16;
      af[mt].q[0] = *(const uint4*)&As[buf][r * LDK + cb];
      af[mt].q[1] = *(const uint4*)&As[buf][r * LDK + cb + 16];
    }
    #pragma unroll
    for (int nt = 0; nt < 4; nt++) {
      int r = (wn << 6) + (nt << 4) + r16;
      bf_[nt].q[0] = *(const uint4*)&Ws[buf][r * LDK + cb];
      bf_[nt].q[1] = *(const uint4*)&Ws[buf][r * LDK + cb + 16];
    }
    #pragma unroll
    for (int mt = 0; mt < 2; mt++)
      #pragma unroll
      for (int nt = 0; nt < 4; nt++)
        acc[mt][nt] = wmma_bf(af[mt], bf_[nt], acc[mt][nt]);
  }

  #pragma unroll
  for (int mt = 0; mt < 2; mt++) {
    #pragma unroll
    for (int nt = 0; nt < 4; nt++) {
      int n = n0 + (wn << 6) + (nt << 4) + r16;
      float bv = bias[n];
      #pragma unroll
      for (int j = 0; j < 8; j++) {
        int m = m0 + (wm << 5) + (mt << 4) + j + cb;   // cb doubles as +8 rows
        float val = acc[mt][nt][j] + bv;
        if (MODE == 0) {
          Cf[(size_t)m * N + n] = val;
        } else if (MODE == 1) {
          Cb[(size_t)m * N + n] = f2bf(fmaxf(val, 0.f));
        } else if (MODE == 3) {
          Cf[(size_t)m * N + n] = val;
          Cb[(size_t)m * N + n] = f2bf(val);
        } else {
          int win = m / S_WIN, s2 = m % S_WIN;
          int seg = n >> 9, h = (n >> 6) & 7, c = n & 63;
          int bh = (win << 3) + h;
          if (seg == 0)      Qb[((size_t)bh * SPAD  + s2) * 64 + c] = f2bf(val * 0.125f);
          else if (seg == 1) Kb[((size_t)bh * S_WIN + s2) * 64 + c] = f2bf(val);
          else               Vb[((size_t)bh * S_WIN + s2) * 64 + c] = f2bf(val);
        }
      }
    }
  }
}

// ---------------------------------------------------------------------------
// Flash attention per (window*head, q-chunk of 128 rows). 8 waves x 16 q rows.
// Transposed formulation: S^T = K * Q^T, O^T = V^T * P^T. In the C layout the
// q index is the lane column, so softmax row stats are per-lane scalars with a
// single shfl_xor(16) per reduction, and P^T in registers IS a valid bf16
// B-fragment (no LDS bounce). K tile staged via async copy; V transposed in
// LDS; O^T transposed back through LDS once at the end. Output bf16.
// ---------------------------------------------------------------------------
__global__ __launch_bounds__(256)
void attn_fa(const unsigned short* __restrict__ Qb,
             const unsigned short* __restrict__ Kb,
             const unsigned short* __restrict__ Vb,
             unsigned short* __restrict__ out, int use_mask)
{
  __shared__ unsigned short kt[32 * 72];     // K tile natural [krow][chan], padded
  __shared__ unsigned short vt[64 * 40];     // V tile transposed [chan][krow], padded
  __shared__ unsigned short ot[8][16 * 72];  // per-wave O^T -> O transpose bounce
  const int bh = blockIdx.x, qc = blockIdx.y;
  const int tid = threadIdx.x, lane = tid & 31, wave = tid >> 5;
  const int win = bh >> 3, hh = bh & 7;
  const int qbase = qc * 128 + wave * 16;
  const int r16 = lane & 15, cb = (lane >> 4) << 3;

  // Q rows as B-fragments (lane column = q row), chan halves 0..31 / 32..63
  FragB qb0, qb1;
  const unsigned short* qp = Qb + ((size_t)bh * SPAD + qbase) * 64;
  qb0.q[0] = *(const uint4*)(qp + (size_t)r16 * 64 + cb);
  qb0.q[1] = *(const uint4*)(qp + (size_t)r16 * 64 + cb + 16);
  qb1.q[0] = *(const uint4*)(qp + (size_t)r16 * 64 + 32 + cb);
  qb1.q[1] = *(const uint4*)(qp + (size_t)r16 * 64 + 32 + cb + 16);

  float m_ = -1e30f, l_ = 0.f;     // per-lane: one q column
  v8f acc[4];                      // O^T tiles: chan = nt*16 + j + cb, q = lane&15
  #pragma unroll
  for (int nt = 0; nt < 4; nt++)
    #pragma unroll
    for (int j = 0; j < 8; j++) acc[nt][j] = 0.f;

  const bool masked = use_mask && ((win % NW) == 0);
  const int qfrm = (qbase + r16) / FQN;            // q frame (pad rows harmless)
  const int rr = tid >> 3, cc = (tid & 7) << 3;

  for (int kk = 0; kk < S_WIN; kk += 32) {
    // K tile: async global->LDS; V tile: transposed store via VGPRs
    async_b128(Kb + ((size_t)bh * S_WIN + kk + rr) * 64 + cc, &kt[rr * 72 + cc]);
    union { uint4 u; unsigned short s[8]; } vv;
    vv.u = *(const uint4*)(Vb + ((size_t)bh * S_WIN + kk + rr) * 64 + cc);
    #pragma unroll
    for (int i = 0; i < 8; i++) vt[(cc + i) * 40 + rr] = vv.s[i];
    wait_async0();
    __syncthreads();

    // S^T tiles: A = K rows, B = Q rows. sc[t][j]: krow = kk+t*16+j+cb, q=lane&15
    v8f sc[2];
    #pragma unroll
    for (int t = 0; t < 2; t++) {
      FragB ka0, ka1;
      int kr = (t << 4) + r16;
      ka0.q[0] = *(const uint4*)&kt[kr * 72 + cb];
      ka0.q[1] = *(const uint4*)&kt[kr * 72 + cb + 16];
      ka1.q[0] = *(const uint4*)&kt[kr * 72 + 32 + cb];
      ka1.q[1] = *(const uint4*)&kt[kr * 72 + 32 + cb + 16];
      v8f z;
      #pragma unroll
      for (int j = 0; j < 8; j++) z[j] = 0.f;
      z = wmma_bf(ka0, qb0, z);
      sc[t] = wmma_bf(ka1, qb1, z);
    }

    if (masked) {
      #pragma unroll
      for (int t = 0; t < 2; t++)
        #pragma unroll
        for (int j = 0; j < 8; j++) {
          int krow = kk + (t << 4) + j + cb;
          bool ck = (krow / FQN) < 4;
          if ((qfrm < 4) != ck) sc[t][j] += -1000.f;
        }
    }

    // online softmax: per-lane register tree + one xor-16 combine
    float mx = fmaxf(sc[0][0], sc[1][0]);
    #pragma unroll
    for (int j = 1; j < 8; j++) mx = fmaxf(mx, fmaxf(sc[0][j], sc[1][j]));
    mx = fmaxf(mx, __shfl_xor(mx, 16, 32));
    float mn = fmaxf(m_, mx);
    float alpha = __expf(m_ - mn);
    float rs = 0.f;
    #pragma unroll
    for (int t = 0; t < 2; t++)
      #pragma unroll
      for (int j = 0; j < 8; j++) {
        float p = __expf(sc[t][j] - mn);
        sc[t][j] = p; rs += p;
      }
    rs += __shfl_xor(rs, 16, 32);
    l_ = l_ * alpha + rs;
    m_ = mn;
    #pragma unroll
    for (int nt = 0; nt < 4; nt++)
      #pragma unroll
      for (int j = 0; j < 8; j++) acc[nt][j] *= alpha;

    // P^T in C layout == bf16 B-fragment over K=32 krows: pure register pack
    FragB pa;
    #pragma unroll
    for (int j = 0; j < 8; j++) {
      pa.s[j]     = f2bf(sc[0][j]);
      pa.s[8 + j] = f2bf(sc[1][j]);
    }
    // O^T += V^T * P^T  (A = V^T rows = channels from vt); preload all frags
    FragB av[4];
    #pragma unroll
    for (int nt = 0; nt < 4; nt++) {
      int nr = (nt << 4) + r16;
      av[nt].q[0] = *(const uint4*)&vt[nr * 40 + cb];
      av[nt].q[1] = *(const uint4*)&vt[nr * 40 + cb + 16];
    }
    #pragma unroll
    for (int nt = 0; nt < 4; nt++)
      acc[nt] = wmma_bf(av[nt], pa, acc[nt]);
    __syncthreads();
  }

  // normalize (per-lane scalar), transpose O^T -> O via LDS, store coalesced
  float inv = 1.f / l_;
  unsigned short* ow = ot[wave];
  #pragma unroll
  for (int nt = 0; nt < 4; nt++)
    #pragma unroll
    for (int j = 0; j < 8; j++) {
      int chan = (nt << 4) + j + cb;
      ow[r16 * 72 + chan] = f2bf(acc[nt][j] * inv);
    }
  const int orow = lane >> 1, ochunk = (lane & 1) << 5;
  const int qrow = qbase + orow;
  if (qrow < S_WIN) {
    uint4 d0 = *(const uint4*)&ow[orow * 72 + ochunk];
    uint4 d1 = *(const uint4*)&ow[orow * 72 + ochunk + 8];
    uint4 d2 = *(const uint4*)&ow[orow * 72 + ochunk + 16];
    uint4 d3 = *(const uint4*)&ow[orow * 72 + ochunk + 24];
    unsigned short* op = out + ((size_t)win * S_WIN + qrow) * DMODEL + hh * 64 + ochunk;
    *(uint4*)(op)      = d0;
    *(uint4*)(op + 8)  = d1;
    *(uint4*)(op + 16) = d2;
    *(uint4*)(op + 24) = d3;
  }
}

// residual add + LayerNorm, one wave per row of 512; dual f32 + bf16 output
__global__ __launch_bounds__(256)
void ln_add(const float* x, const float* y,
            const float* __restrict__ sg, const float* __restrict__ bg,
            float* out, unsigned short* outb)
{
  const int row  = blockIdx.x * 8 + (threadIdx.x >> 5);
  const int lane = threadIdx.x & 31;
  const float* xr = x + (size_t)row * DMODEL;
  const float* yr = y + (size_t)row * DMODEL;
  float v[16]; float sum = 0.f;
  #pragma unroll
  for (int i = 0; i < 16; i++) { int c = lane + (i << 5); v[i] = xr[c] + yr[c]; sum += v[i]; }
  for (int o = 1; o < 32; o <<= 1) sum += __shfl_xor(sum, o, 32);
  float mu = sum * (1.0f / DMODEL);
  float var = 0.f;
  #pragma unroll
  for (int i = 0; i < 16; i++) { float d = v[i] - mu; var += d * d; }
  for (int o = 1; o < 32; o <<= 1) var += __shfl_xor(var, o, 32);
  float rstd = rsqrtf(var * (1.0f / DMODEL) + 1e-5f);
  float* orow = out + (size_t)row * DMODEL;
  unsigned short* brow = outb + (size_t)row * DMODEL;
  #pragma unroll
  for (int i = 0; i < 16; i++) {
    int c = lane + (i << 5);
    float r = (v[i] - mu) * rstd * sg[c] + bg[c];
    orow[c] = r;
    brow[c] = f2bf(r);
  }
}

// roll along T (frames): out[b,t] = in[b,(t-shift+T)%T]; dual f32 + bf16 output
__global__ void roll_t(const float* __restrict__ in, float* __restrict__ out,
                       unsigned short* __restrict__ outb, int shift)
{
  size_t i = (size_t)blockIdx.x * 256 + threadIdx.x;   // ROWS*128 float4 units
  int c4 = (int)(i & 127);
  size_t rowi = i >> 7;
  int b   = (int)(rowi / (T_FRAMES * FQN));
  int rem = (int)(rowi % (T_FRAMES * FQN));
  int t = rem / FQN, fq = rem % FQN;
  int ts = (t - shift + T_FRAMES) % T_FRAMES;
  size_t src = ((size_t)b * T_FRAMES * FQN + (size_t)ts * FQN + fq) * DMODEL + (size_t)c4 * 4;
  float4 v = *(const float4*)(in + src);
  *(float4*)(out + rowi * DMODEL + (size_t)c4 * 4) = v;
  uint2 p; p.x = pack2(v.x, v.y); p.y = pack2(v.z, v.w);
  *(uint2*)(outb + rowi * DMODEL + (size_t)c4 * 4) = p;
}

// fp32 -> bf16 bulk convert (n multiple of 4)
__global__ void cvt_bf16(const float* __restrict__ src, unsigned short* __restrict__ dst, long n)
{
  long i = ((long)blockIdx.x * 256 + threadIdx.x) * 4;
  if (i < n) {
    float4 v = *(const float4*)(src + i);
    uint2 p; p.x = pack2(v.x, v.y); p.y = pack2(v.z, v.w);
    *(uint2*)(dst + i) = p;
  }
}

__global__ void qpad_zero(unsigned short* Qb)
{
  int i = blockIdx.x * 256 + threadIdx.x;
  const int total = BHN * (SPAD - S_WIN) * 64;
  if (i < total) {
    int bh = i / ((SPAD - S_WIN) * 64);
    int r  = i % ((SPAD - S_WIN) * 64);
    int s = S_WIN + r / 64, c = r % 64;
    Qb[((size_t)bh * SPAD + s) * 64 + c] = 0;
  }
}

extern "C" void kernel_launch(void* const* d_in, const int* in_sizes, int n_in,
                              void* d_out, int out_size, void* d_ws, size_t ws_size,
                              hipStream_t stream)
{
  (void)in_sizes; (void)n_in; (void)out_size; (void)ws_size;
  const float* fqin = (const float*)d_in[0];
  const float* Wp   = (const float*)d_in[1];
  const float* bp   = (const float*)d_in[2];
  const float* Wqkv = (const float*)d_in[3];
  const float* bqkv = (const float*)d_in[4];
  const float* Wo   = (const float*)d_in[5];
  const float* bo   = (const float*)d_in[6];
  const float* ln1s = (const float*)d_in[7];
  const float* ln1b = (const float*)d_in[8];
  const float* W1   = (const float*)d_in[9];
  const float* b1   = (const float*)d_in[10];
  const float* W2   = (const float*)d_in[11];
  const float* b2   = (const float*)d_in[12];
  const float* ln2s = (const float*)d_in[13];
  const float* ln2b = (const float*)d_in[14];
  float* x = (float*)d_out;   // running activation [9600,512], final output

  char* w = (char*)d_ws;
  auto carve = [&](size_t bytes) -> void* {
    void* p = (void*)w; w += (bytes + 255) & ~(size_t)255; return p;
  };
  float*          xroll    = (float*)carve((size_t)ROWS * DMODEL * 4);
  float*          proj     = (float*)carve((size_t)ROWS * DMODEL * 4);
  unsigned short* x_bf     = (unsigned short*)carve((size_t)ROWS * DMODEL * 2);
  unsigned short* xroll_bf = (unsigned short*)carve((size_t)ROWS * DMODEL * 2);
  unsigned short* fq_bf    = (unsigned short*)carve((size_t)ROWS * DMODEL * 2);
  unsigned short* attn_bf  = (unsigned short*)carve((size_t)ROWS * DMODEL * 2);
  unsigned short* ffnh_bf  = (unsigned short*)carve((size_t)ROWS * DFF * 2);
  unsigned short* Qb = (unsigned short*)carve((size_t)BHN * SPAD  * 64 * 2);
  unsigned short* Kb = (unsigned short*)carve((size_t)BHN * S_WIN * 64 * 2);
  unsigned short* Vb = (unsigned short*)carve((size_t)BHN * S_WIN * 64 * 2);
  unsigned short* Wp_b   = (unsigned short*)carve((size_t)DMODEL * DMODEL * 2);
  unsigned short* Wqkv_b = (unsigned short*)carve((size_t)NLAYER * 3 * DMODEL * DMODEL * 2);
  unsigned short* Wo_b   = (unsigned short*)carve((size_t)NLAYER * DMODEL * DMODEL * 2);
  unsigned short* W1_b   = (unsigned short*)carve((size_t)NLAYER * DFF * DMODEL * 2);
  unsigned short* W2_b   = (unsigned short*)carve((size_t)NLAYER * DMODEL * DFF * 2);

  dim3 blk(256);
  auto cvt = [&](const float* s, unsigned short* d, long n) {
    cvt_bf16<<<(unsigned)((n / 4 + 255) / 256), blk, 0, stream>>>(s, d, n);
  };
  cvt(fqin, fq_bf, (long)ROWS * DMODEL);
  cvt(Wp,   Wp_b,  (long)DMODEL * DMODEL);
  cvt(Wqkv, Wqkv_b,(long)NLAYER * 3 * DMODEL * DMODEL);
  cvt(Wo,   Wo_b,  (long)NLAYER * DMODEL * DMODEL);
  cvt(W1,   W1_b,  (long)NLAYER * DFF * DMODEL);
  cvt(W2,   W2_b,  (long)NLAYER * DMODEL * DFF);
  qpad_zero<<<(BHN * (SPAD - S_WIN) * 64 + 255) / 256, blk, 0, stream>>>(Qb);

  // x = frame_queries @ Wp^T + bp   (f32 + bf16 copies)
  gemm_bf16<3><<<dim3(DMODEL / BN, ROWS / BM), blk, 0, stream>>>(
      fq_bf, Wp_b, bp, x, x_bf, nullptr, nullptr, nullptr, ROWS, DMODEL, DMODEL);

  for (int i = 0; i < NLAYER; i++) {
    const int odd = i & 1;
    float* xio = odd ? xroll : x;
    unsigned short* xio_bf = odd ? xroll_bf : x_bf;
    if (odd) roll_t<<<(ROWS * 128) / 256, blk, 0, stream>>>(x, xroll, xroll_bf, 4);

    // QKV GEMM, epilogue packs bf16 per-head Q(x0.125)/K/V
    gemm_bf16<2><<<dim3((3 * DMODEL) / BN, ROWS / BM), blk, 0, stream>>>(
        xio_bf, Wqkv_b + (size_t)i * 3 * DMODEL * DMODEL, bqkv + (size_t)i * 3 * DMODEL,
        nullptr, nullptr, Qb, Kb, Vb, ROWS, 3 * DMODEL, DMODEL);

    attn_fa<<<dim3(BHN, SPAD / 128), blk, 0, stream>>>(Qb, Kb, Vb, attn_bf, odd);

    gemm_bf16<0><<<dim3(DMODEL / BN, ROWS / BM), blk, 0, stream>>>(
        attn_bf, Wo_b + (size_t)i * DMODEL * DMODEL, bo + (size_t)i * DMODEL, proj,
        nullptr, nullptr, nullptr, nullptr, ROWS, DMODEL, DMODEL);

    ln_add<<<ROWS / 8, blk, 0, stream>>>(
        xio, proj, ln1s + (size_t)i * DMODEL, ln1b + (size_t)i * DMODEL, xio, xio_bf);

    if (odd) roll_t<<<(ROWS * 128) / 256, blk, 0, stream>>>(xroll, x, x_bf, -4);

    gemm_bf16<1><<<dim3(DFF / BN, ROWS / BM), blk, 0, stream>>>(
        x_bf, W1_b + (size_t)i * DFF * DMODEL, b1 + (size_t)i * DFF,
        nullptr, ffnh_bf, nullptr, nullptr, nullptr, ROWS, DFF, DMODEL);

    gemm_bf16<0><<<dim3(DMODEL / BN, ROWS / BM), blk, 0, stream>>>(
        ffnh_bf, W2_b + (size_t)i * DMODEL * DFF, b2 + (size_t)i * DMODEL, proj,
        nullptr, nullptr, nullptr, nullptr, ROWS, DMODEL, DFF);

    ln_add<<<ROWS / 8, blk, 0, stream>>>(
        x, proj, ln2s + (size_t)i * DMODEL, ln2b + (size_t)i * DMODEL, x, x_bf);
  }
}
